// MartinezGRU_61083024883781
// MI455X (gfx1250) — matrix operand order repaired and checked by reading, not compiled, not run on hardware
//
#include <hip/hip_runtime.h>
#include <hip/hip_bf16.h>
#include <stdint.h>

// ---------------------------------------------------------------------------
// MartinezGRU seq2seq on MI455X (gfx1250, wave32, WMMA).
//   B=1024 T=100 P=50 I=96 H=1024
// Strategy: bf16 WMMA (16x16x32, fp32 accum) GEMMs for all matmuls.
//   - Weights repacked once into WMMA B-fragment layout (6MB/matrix, L2-resident).
//   - Hidden states kept fp32 + bf16 mirror for the next step's A-matrix.
//   - GEMM K-loop is software-pipelined AND unrolled x2 with ping-pong register
//     buffers: no inter-buffer v_mov copies, loads for block k+1/k+2 in flight
//     while WMMAs consume block k.
//   - Per-step: gi/gh GEMMs -> fused GRU pointwise; decoder adds fc_in/fc_out
//     GEMMs, fc_out writes residual result directly into d_out (strided).
// ---------------------------------------------------------------------------

#define BB 1024
#define TT 100
#define PP 50
#define II 96
#define HH 1024
#define H3 (3 * HH)

typedef __bf16 v16bf __attribute__((ext_vector_type(16)));
typedef __bf16 v8bf  __attribute__((ext_vector_type(8)));
typedef float  v8f   __attribute__((ext_vector_type(8)));

__device__ __forceinline__ unsigned short f2bf(float f) {
  // round-to-nearest-even fp32 -> bf16
  unsigned u = __float_as_uint(f);
  unsigned r = u + 0x7FFFu + ((u >> 16) & 1u);
  return (unsigned short)(r >> 16);
}

__device__ __forceinline__ v16bf mergeA(v8bf lo, v8bf hi) {
  return __builtin_shufflevector(lo, hi, 0, 1, 2, 3, 4, 5, 6, 7, 8, 9, 10, 11,
                                 12, 13, 14, 15);
}

// ---------------------------------------------------------------------------
// Repack fp32 weight W (N x K, row-major, out = x @ W^T) into a WMMA
// 16-bit B-matrix (32x16 KxN) fragment layout in contiguous k order (matches the A loads):
//   Pk[((ntile*(K/32) + kblk)*32 + lane)*16 + i] = bf16(W[ntile*16 + (lane&15)]
//                                                       [kblk*32 + (lane>>4)*16 + i])
// One thread emits one lane's 16 contiguous bf16 values (32B).
// ---------------------------------------------------------------------------
__global__ void pack_weight_bf16(const float* __restrict__ W,
                                 unsigned short* __restrict__ Pk,
                                 int N, int K) {
  int tid = blockIdx.x * blockDim.x + threadIdx.x;
  int kblks = K >> 5;
  int total = (N >> 4) * kblks * 32;
  if (tid >= total) return;
  int lane = tid & 31;
  int rest = tid >> 5;
  int kblk = rest % kblks;
  int ntile = rest / kblks;
  int n = ntile * 16 + (lane & 15);
  int kbase = kblk * 32 + ((lane >> 4) << 4);
  const float* src = W + (size_t)n * K + kbase;
  unsigned short* dst = Pk + (size_t)tid * 16;
#pragma unroll
  for (int i = 0; i < 16; ++i) dst[i] = f2bf(src[i]);
}

// ---------------------------------------------------------------------------
// C(M x N) = A(M x K bf16 row-major, leading dim lda) @ Bpacked
//            [+ bias] [+ residual] -> optional fp32 out, optional bf16 out.
// Block = 128 threads = 4 waves; wave w owns M-tile (blockIdx.y*4+w),
// each wave computes a 16 x (16*NT) strip via v_wmma_f32_16x16x32_bf16.
// K-loop: unrolled x2 with ping-pong register buffers (no rotation movs);
// loads for the next K-block issue before the WMMAs consuming the current one.
// Requires: M % 64 == 0, N % (16*NT) == 0, K % 32 == 0.
// No divergence anywhere (EXEC must be all 1s for WMMA).
// ---------------------------------------------------------------------------
template <int NT>
__global__ __launch_bounds__(128)
void gemm_bf16_wmma(const unsigned short* __restrict__ A, int lda,
                    const unsigned short* __restrict__ Bp,
                    const float* __restrict__ bias,
                    const float* __restrict__ residual, long long ldr,
                    float* __restrict__ Cf, long long ldc,
                    unsigned short* __restrict__ Cbf, int ldcb,
                    int K) {
  const int lane = threadIdx.x & 31;
  const int wave = threadIdx.x >> 5;
  const int m0 = (blockIdx.y * 4 + wave) * 16;
  const int nt0 = blockIdx.x * NT;  // first 16-wide N tile index
  const int kblks = K >> 5;

  const int sel = lane >> 4;          // half-wave select
  const int mrow = m0 + (lane & 15);  // A row for this lane

  const unsigned short* arow = A + (size_t)mrow * lda + sel * 16;
  const unsigned short* bptr = Bp + ((size_t)nt0 * kblks * 32 + lane) * 16;
  const size_t bstride = (size_t)kblks * 512;  // elements per packed N-tile

  v8f acc[NT] = {};

  v8bf alo0, ahi0, alo1, ahi1;
  v16bf b0[NT], b1[NT];

  // ---- prologue: K-block 0 -> buffer 0 ----
  alo0 = *(const v8bf*)(arow);
  ahi0 = *(const v8bf*)(arow + 8);
#pragma unroll
  for (int i = 0; i < NT; ++i) b0[i] = *(const v16bf*)(bptr + i * bstride);

  // ---- steady state: 2 K-blocks per iteration, ping-pong buffers ----
  int kb = 0;
  for (; kb + 2 < kblks; kb += 2) {
    // load block kb+1 -> buffer 1
    {
      const unsigned short* a1 = arow + 32;
      const unsigned short* p1 = bptr + 512;
      alo1 = *(const v8bf*)(a1);
      ahi1 = *(const v8bf*)(a1 + 8);
#pragma unroll
      for (int i = 0; i < NT; ++i) b1[i] = *(const v16bf*)(p1 + i * bstride);
      __builtin_prefetch(p1 + 512, 0, 3);  // speculative: fault-free per ISA
    }
    // multiply block kb (buffer 0)
    {
      v16bf a = mergeA(alo0, ahi0);
#pragma unroll
      for (int i = 0; i < NT; ++i)
        acc[i] = __builtin_amdgcn_wmma_f32_16x16x32_bf16(
            false, a, false, b0[i], (short)0, acc[i], false, false);
    }
    // load block kb+2 -> buffer 0
    arow += 64;
    bptr += 1024;
    {
      alo0 = *(const v8bf*)(arow);
      ahi0 = *(const v8bf*)(arow + 8);
#pragma unroll
      for (int i = 0; i < NT; ++i) b0[i] = *(const v16bf*)(bptr + i * bstride);
      __builtin_prefetch(bptr + 512, 0, 3);
    }
    // multiply block kb+1 (buffer 1)
    {
      v16bf a = mergeA(alo1, ahi1);
#pragma unroll
      for (int i = 0; i < NT; ++i)
        acc[i] = __builtin_amdgcn_wmma_f32_16x16x32_bf16(
            false, a, false, b1[i], (short)0, acc[i], false, false);
    }
  }

  // ---- tail: 1 or 2 blocks remain; buffer 0 holds block kb ----
  if (kb + 1 < kblks) {
    const unsigned short* a1 = arow + 32;
    const unsigned short* p1 = bptr + 512;
    alo1 = *(const v8bf*)(a1);
    ahi1 = *(const v8bf*)(a1 + 8);
#pragma unroll
    for (int i = 0; i < NT; ++i) b1[i] = *(const v16bf*)(p1 + i * bstride);
    v16bf a = mergeA(alo0, ahi0);
#pragma unroll
    for (int i = 0; i < NT; ++i)
      acc[i] = __builtin_amdgcn_wmma_f32_16x16x32_bf16(
          false, a, false, b0[i], (short)0, acc[i], false, false);
    a = mergeA(alo1, ahi1);
#pragma unroll
    for (int i = 0; i < NT; ++i)
      acc[i] = __builtin_amdgcn_wmma_f32_16x16x32_bf16(
          false, a, false, b1[i], (short)0, acc[i], false, false);
  } else {
    v16bf a = mergeA(alo0, ahi0);
#pragma unroll
    for (int i = 0; i < NT; ++i)
      acc[i] = __builtin_amdgcn_wmma_f32_16x16x32_bf16(
          false, a, false, b0[i], (short)0, acc[i], false, false);
  }

  // C/D layout: lane holds column n, VGPR r holds row m0 + sel*8 + r
  const int nbase = nt0 * 16 + (lane & 15);
#pragma unroll
  for (int i = 0; i < NT; ++i) {
    const int n = nbase + i * 16;
    const float bv = bias ? bias[n] : 0.0f;
#pragma unroll
    for (int r = 0; r < 8; ++r) {
      const int m = m0 + sel * 8 + r;
      float v = acc[i][r] + bv;
      if (residual) v += residual[(size_t)m * ldr + n];
      if (Cf) Cf[(size_t)m * ldc + n] = v;
      if (Cbf) Cbf[(size_t)m * ldcb + n] = f2bf(v);
    }
  }
}

// ---------------------------------------------------------------------------
// Fused GRU gate math: gi/gh are (B x 3H) with biases already added by GEMM.
//   r = sig(ir+hr); z = sig(iz+hz); n = tanh(in + r*hn); h' = (1-z)*n + z*h
// Updates fp32 state in place + bf16 mirror.
// ---------------------------------------------------------------------------
__global__ void gru_pointwise(const float* __restrict__ gi,
                              const float* __restrict__ gh,
                              float* __restrict__ hf,
                              unsigned short* __restrict__ hb) {
  int idx = blockIdx.x * blockDim.x + threadIdx.x;
  if (idx >= BB * HH) return;
  int b = idx >> 10;       // / HH
  int j = idx & (HH - 1);  // % HH
  size_t base = (size_t)b * H3 + j;
  float ir = gi[base], iz = gi[base + HH], inn = gi[base + 2 * HH];
  float hr = gh[base], hz = gh[base + HH], hn = gh[base + 2 * HH];
  float h = hf[idx];
  float r = 1.0f / (1.0f + __expf(-(ir + hr)));
  float z = 1.0f / (1.0f + __expf(-(iz + hz)));
  float n = tanhf(inn + r * hn);
  float out = (1.0f - z) * n + z * h;
  hf[idx] = out;
  hb[idx] = f2bf(out);
}

__global__ void cast_f32_to_bf16(const float* __restrict__ src,
                                 unsigned short* __restrict__ dst, int n) {
  int i = blockIdx.x * blockDim.x + threadIdx.x;
  if (i < n) dst[i] = f2bf(src[i]);
}

__global__ void zero_state(float* __restrict__ hf,
                           unsigned short* __restrict__ hb, int n) {
  int i = blockIdx.x * blockDim.x + threadIdx.x;
  if (i < n) {
    hf[i] = 0.0f;
    hb[i] = 0;
  }
}

// out[:, :P, :] = x[:, :P, :]
__global__ void copy_prefix(const float* __restrict__ x,
                            float* __restrict__ out) {
  int i = blockIdx.x * blockDim.x + threadIdx.x;
  const int n = BB * PP * II;
  if (i >= n) return;
  int b = i / (PP * II);
  int r = i % (PP * II);
  size_t off = (size_t)b * TT * II + r;
  out[off] = x[off];
}

// frame_bf = bf16(x[:, P-1, :])
__global__ void init_frame(const float* __restrict__ x,
                           unsigned short* __restrict__ fb) {
  int i = blockIdx.x * blockDim.x + threadIdx.x;
  if (i >= BB * II) return;
  int b = i / II;
  int c = i % II;
  fb[i] = f2bf(x[(size_t)b * TT * II + (size_t)(PP - 1) * II + c]);
}

// ---------------------------------------------------------------------------
extern "C" void kernel_launch(void* const* d_in, const int* in_sizes, int n_in,
                              void* d_out, int out_size, void* d_ws,
                              size_t ws_size, hipStream_t stream) {
  const float* x = (const float*)d_in[0];
  const float* eWih0 = (const float*)d_in[1];
  const float* eWhh0 = (const float*)d_in[2];
  const float* ebih0 = (const float*)d_in[3];
  const float* ebhh0 = (const float*)d_in[4];
  const float* eWih1 = (const float*)d_in[5];
  const float* eWhh1 = (const float*)d_in[6];
  const float* ebih1 = (const float*)d_in[7];
  const float* ebhh1 = (const float*)d_in[8];
  const float* dWih0 = (const float*)d_in[9];
  const float* dWhh0 = (const float*)d_in[10];
  const float* dbih0 = (const float*)d_in[11];
  const float* dbhh0 = (const float*)d_in[12];
  const float* dWih1 = (const float*)d_in[13];
  const float* dWhh1 = (const float*)d_in[14];
  const float* dbih1 = (const float*)d_in[15];
  const float* dbhh1 = (const float*)d_in[16];
  const float* fc_in_w = (const float*)d_in[17];
  const float* fc_in_b = (const float*)d_in[18];
  const float* fc_out_w = (const float*)d_in[19];
  const float* fc_out_b = (const float*)d_in[20];
  float* out = (float*)d_out;

  // ---- workspace carve-up (256B aligned) ----
  char* ws = (char*)d_ws;
  size_t off = 0;
  auto alloc = [&](size_t bytes) -> void* {
    void* p = ws + off;
    off = (off + bytes + 255) & ~(size_t)255;
    return p;
  };
  auto packBytes = [](int N, int K) -> size_t {
    return (size_t)(N >> 4) * (K >> 5) * 512 * sizeof(unsigned short);
  };

  unsigned short* x_bf = (unsigned short*)alloc((size_t)BB * TT * II * 2);
  unsigned short* p_eWih0 = (unsigned short*)alloc(packBytes(H3, II));
  unsigned short* p_eWhh0 = (unsigned short*)alloc(packBytes(H3, HH));
  unsigned short* p_eWih1 = (unsigned short*)alloc(packBytes(H3, HH));
  unsigned short* p_eWhh1 = (unsigned short*)alloc(packBytes(H3, HH));
  unsigned short* p_dWih0 = (unsigned short*)alloc(packBytes(H3, HH));
  unsigned short* p_dWhh0 = (unsigned short*)alloc(packBytes(H3, HH));
  unsigned short* p_dWih1 = (unsigned short*)alloc(packBytes(H3, HH));
  unsigned short* p_dWhh1 = (unsigned short*)alloc(packBytes(H3, HH));
  unsigned short* p_fcin = (unsigned short*)alloc(packBytes(HH, II));
  unsigned short* p_fcout = (unsigned short*)alloc(packBytes(II, HH));
  float* gi = (float*)alloc((size_t)BB * H3 * 4);
  float* gh = (float*)alloc((size_t)BB * H3 * 4);
  float* h0f = (float*)alloc((size_t)BB * HH * 4);
  float* h1f = (float*)alloc((size_t)BB * HH * 4);
  unsigned short* h0b = (unsigned short*)alloc((size_t)BB * HH * 2);
  unsigned short* h1b = (unsigned short*)alloc((size_t)BB * HH * 2);
  unsigned short* y_bf = (unsigned short*)alloc((size_t)BB * HH * 2);
  unsigned short* frame_bf = (unsigned short*)alloc((size_t)BB * II * 2);
  if (off > ws_size) return;  // insufficient scratch; avoid corrupting memory

  const int TPB = 256;
  auto blocks = [](int n) { return (n + 255) / 256; };

  // NT=4 (16x64 strip/wave) when N % 64 == 0, else NT=2 (fc_out N=96).
  auto gemm = [&](const unsigned short* A, int lda, const unsigned short* Bp,
                  const float* bias, const float* res, long long ldr, float* Cf,
                  long long ldc, unsigned short* Cbf, int ldcb, int N, int K) {
    dim3 b(128);
    if ((N & 63) == 0) {
      dim3 g(N / 64, BB / 64);
      gemm_bf16_wmma<4><<<g, b, 0, stream>>>(A, lda, Bp, bias, res, ldr, Cf,
                                             ldc, Cbf, ldcb, K);
    } else {
      dim3 g(N / 32, BB / 64);
      gemm_bf16_wmma<2><<<g, b, 0, stream>>>(A, lda, Bp, bias, res, ldr, Cf,
                                             ldc, Cbf, ldcb, K);
    }
  };
  auto pack = [&](const float* W, unsigned short* Pk, int N, int K) {
    int total = (N >> 4) * (K >> 5) * 32;
    pack_weight_bf16<<<blocks(total), TPB, 0, stream>>>(W, Pk, N, K);
  };

  // ---- one-time prep (per launch; deterministic) ----
  cast_f32_to_bf16<<<blocks(BB * TT * II), TPB, 0, stream>>>(x, x_bf,
                                                             BB * TT * II);
  pack(eWih0, p_eWih0, H3, II);
  pack(eWhh0, p_eWhh0, H3, HH);
  pack(eWih1, p_eWih1, H3, HH);
  pack(eWhh1, p_eWhh1, H3, HH);
  pack(dWih0, p_dWih0, H3, HH);
  pack(dWhh0, p_dWhh0, H3, HH);
  pack(dWih1, p_dWih1, H3, HH);
  pack(dWhh1, p_dWhh1, H3, HH);
  pack(fc_in_w, p_fcin, HH, II);
  pack(fc_out_w, p_fcout, II, HH);
  zero_state<<<blocks(BB * HH), TPB, 0, stream>>>(h0f, h0b, BB * HH);
  zero_state<<<blocks(BB * HH), TPB, 0, stream>>>(h1f, h1b, BB * HH);
  copy_prefix<<<blocks(BB * PP * II), TPB, 0, stream>>>(x, out);

  // ---- encoder: 2 stacked GRU layers, interleaved per timestep ----
  for (int t = 0; t < PP; ++t) {
    gemm(x_bf + (size_t)t * II, TT * II, p_eWih0, ebih0, nullptr, 0, gi, H3,
         nullptr, 0, H3, II);
    gemm(h0b, HH, p_eWhh0, ebhh0, nullptr, 0, gh, H3, nullptr, 0, H3, HH);
    gru_pointwise<<<blocks(BB * HH), TPB, 0, stream>>>(gi, gh, h0f, h0b);
    gemm(h0b, HH, p_eWih1, ebih1, nullptr, 0, gi, H3, nullptr, 0, H3, HH);
    gemm(h1b, HH, p_eWhh1, ebhh1, nullptr, 0, gh, H3, nullptr, 0, H3, HH);
    gru_pointwise<<<blocks(BB * HH), TPB, 0, stream>>>(gi, gh, h1f, h1b);
  }

  // ---- decoder: autoregressive; fc_out writes straight into d_out ----
  init_frame<<<blocks(BB * II), TPB, 0, stream>>>(x, frame_bf);
  for (int s = 0; s < TT - PP; ++s) {
    gemm(frame_bf, II, p_fcin, fc_in_b, nullptr, 0, nullptr, 0, y_bf, HH, HH,
         II);
    gemm(y_bf, HH, p_dWih0, dbih0, nullptr, 0, gi, H3, nullptr, 0, H3, HH);
    gemm(h0b, HH, p_dWhh0, dbhh0, nullptr, 0, gh, H3, nullptr, 0, H3, HH);
    gru_pointwise<<<blocks(BB * HH), TPB, 0, stream>>>(gi, gh, h0f, h0b);
    gemm(h0b, HH, p_dWih1, dbih1, nullptr, 0, gi, H3, nullptr, 0, H3, HH);
    gemm(h1b, HH, p_dWhh1, dbhh1, nullptr, 0, gh, H3, nullptr, 0, H3, HH);
    gru_pointwise<<<blocks(BB * HH), TPB, 0, stream>>>(gi, gh, h1f, h1b);
    // frame = h1 @ fc_out^T + b + prev_frame; prev frame lives in d_out
    gemm(h1b, HH, p_fcout, fc_out_b, out + (size_t)(PP + s - 1) * II,
         (long long)TT * II, out + (size_t)(PP + s) * II, (long long)TT * II,
         frame_bf, II, II, HH);
  }
}
